// Seq2Seq_62740882260290
// MI455X (gfx1250) — matrix operand order repaired and checked by reading, not compiled, not run on hardware
//
#include <hip/hip_runtime.h>

#define B_  64
#define S_  64
#define T_  32
#define E_  512
#define U_  1024
#define VD_ 32000
#define FU_ 4096   /* 4*U */
#define TU_ 2048   /* 2*U */

typedef __bf16 bf16;
typedef bf16  v16bf __attribute__((ext_vector_type(16)));
typedef bf16  v8bf  __attribute__((ext_vector_type(8)));
typedef float v8f   __attribute__((ext_vector_type(8)));

union FragA { v16bf v; v8bf h[2]; };

__device__ __forceinline__ unsigned short f2bf(float f) {
    unsigned int u = __float_as_uint(f);
    if ((u & 0x7fffffffu) > 0x7f800000u) return (unsigned short)((u >> 16) | 0x40); // NaN
    u += 0x7fffu + ((u >> 16) & 1u);   // round-to-nearest-even
    return (unsigned short)(u >> 16);
}
__device__ __forceinline__ float sigm(float x) { return 1.0f / (1.0f + expf(-x)); }

// ---------------------------------------------------------------- casts / gathers
__global__ void castk(const float* __restrict__ s, unsigned short* __restrict__ d, long long n) {
    long long i = (long long)blockIdx.x * blockDim.x + threadIdx.x;
    if (i < n) d[i] = f2bf(s[i]);
}

__global__ void gather_emb(const int* __restrict__ tok, int rowsPer, int tokStride,
                           const float* __restrict__ emb, unsigned short* __restrict__ dst,
                           long long total) {
    long long i = (long long)blockIdx.x * blockDim.x + threadIdx.x;
    if (i >= total) return;
    long long rw = i >> 9;               // / E_
    int e = (int)(i & (E_ - 1));
    int bb = (int)(rw / rowsPer), rr = (int)(rw % rowsPer);
    int t = tok[bb * tokStride + rr];
    dst[i] = f2bf(emb[(long long)t * E_ + e]);
}

__global__ void init_state(unsigned short* __restrict__ h_bf, float* __restrict__ c, int n) {
    int i = blockIdx.x * blockDim.x + threadIdx.x;
    if (i < n) { h_bf[i] = 0; c[i] = 0.0f; }
}

// ---------------------------------------------------------------- generic WMMA GEMM
// out[m, n] = sum_k A[m,k] * W[n,k]  (+ bias0[n] + bias1[n] + Z[m*zRowStride + n])
// A, W bf16 row-major; out fp32 with arbitrary row/col strides.
// wave tile: 16 (M) x 64 (N); K stepped by 32 via v_wmma_f32_16x16x32_bf16.
__global__ void __launch_bounds__(256)
wmma_gemm_bf16(const unsigned short* __restrict__ A, int lda,
               const unsigned short* __restrict__ W, int ldw,
               const float* __restrict__ Z, long long zRowStride,
               const float* __restrict__ bias0, const float* __restrict__ bias1,
               float* __restrict__ out, long long oRowStride, long long oColStride,
               int M, long long N, int K)
{
    const int tid  = threadIdx.x;
    const int lane = tid & 31;
    const int wib  = tid >> 5;
    const int wavesM = M >> 4;
    const long long w = (long long)blockIdx.x * (blockDim.x >> 5) + wib;
    const int m0       = (int)(w % wavesM) << 4;
    const long long n0 = (w / wavesM) << 6;
    if (n0 >= N) return;

    const int half = lane >> 4;     // lane group 0..15 / 16..31
    const int r    = lane & 15;

    const unsigned short* arow = A + (long long)(m0 + r) * lda;

    v8f acc[4];
#pragma unroll
    for (int t = 0; t < 4; ++t) {
        const long long n = n0 + t * 16 + r;
        float bz = 0.0f;
        if (bias0) bz += bias0[n];
        if (bias1) bz += bias1[n];
#pragma unroll
        for (int j = 0; j < 8; ++j) {
            const int m = m0 + j + (half << 3);
            acc[t][j] = bz + (Z ? Z[(long long)m * zRowStride + n] : 0.0f);
        }
    }

    for (int kk = 0; kk < K; kk += 32) {
        // A fragment (16x32): lane holds row m = r, K = half*16..half*16+15 contiguous
        FragA a;
        a.h[0] = *(const v8bf*)(arow + kk + half * 16);
        a.h[1] = *(const v8bf*)(arow + kk + half * 16 + 8);
#pragma unroll
        for (int t = 0; t < 4; ++t) {
            // B fragment (32x16): lane holds column n = r, K = half*16..half*16+15 contiguous
            const v16bf bfrag =
                *(const v16bf*)(W + (n0 + t * 16 + r) * (long long)ldw + kk + half * 16);
            acc[t] = __builtin_amdgcn_wmma_f32_16x16x32_bf16(
                false, a.v, false, bfrag, (short)0, acc[t], false, false);
        }
    }

#pragma unroll
    for (int t = 0; t < 4; ++t) {
        const long long n = n0 + t * 16 + r;
#pragma unroll
        for (int j = 0; j < 8; ++j) {
            const int m = m0 + j + (half << 3);
            out[(long long)m * oRowStride + n * oColStride] = acc[t][j];
        }
    }
}

// ---------------------------------------------------------------- LSTM gate math
__global__ void lstm_gates_enc(const float* __restrict__ z, float* __restrict__ c,
                               unsigned short* __restrict__ h_bf, float* __restrict__ o_buf,
                               int t) {
    int idx = blockIdx.x * blockDim.x + threadIdx.x;
    if (idx >= B_ * U_) return;
    int b = idx >> 10, u = idx & (U_ - 1);
    const float* zb = z + (long long)b * FU_;
    float zi = zb[u], zf = zb[U_ + u], zg = zb[2 * U_ + u], zo = zb[3 * U_ + u];
    float cv = sigm(zf) * c[idx] + sigm(zi) * tanhf(zg);
    float hv = sigm(zo) * tanhf(cv);
    c[idx] = cv;
    h_bf[idx] = f2bf(hv);
    o_buf[((long long)b * S_ + t) * U_ + u] = hv;
}

__global__ void lstm_gates_dec(const float* __restrict__ z, float* __restrict__ c,
                               unsigned short* __restrict__ h_bf,
                               unsigned short* __restrict__ hc_bf) {
    int idx = blockIdx.x * blockDim.x + threadIdx.x;
    if (idx >= B_ * U_) return;
    int b = idx >> 10, u = idx & (U_ - 1);
    const float* zb = z + (long long)b * FU_;
    float zi = zb[u], zf = zb[U_ + u], zg = zb[2 * U_ + u], zo = zb[3 * U_ + u];
    float cv = sigm(zf) * c[idx] + sigm(zi) * tanhf(zg);
    float hv = sigm(zo) * tanhf(cv);
    c[idx] = cv;
    unsigned short hb = f2bf(hv);
    h_bf[idx] = hb;
    hc_bf[(long long)b * TU_ + U_ + u] = hb;   // h half of [ctx, h]
}

// ---------------------------------------------------------------- attention (per-batch block)
__global__ void __launch_bounds__(256)
attn_kernel(const float* __restrict__ q, const float* __restrict__ o,
            unsigned short* __restrict__ hc_bf) {
    __shared__ float qs[U_];
    __shared__ float sc[S_];
    __shared__ float ex[S_];
    __shared__ float denom;
    const int b = blockIdx.x, tid = threadIdx.x;
    for (int u = tid; u < U_; u += 256) qs[u] = q[b * U_ + u];
    __syncthreads();
    const int wv = tid >> 5, ln = tid & 31;
    for (int s = wv; s < S_; s += 8) {
        const float* orow = o + ((long long)b * S_ + s) * U_;
        float a = 0.0f;
        for (int u = ln; u < U_; u += 32) a += qs[u] * orow[u];
#pragma unroll
        for (int offm = 16; offm > 0; offm >>= 1) a += __shfl_xor(a, offm, 32);
        if (ln == 0) sc[s] = a;
    }
    __syncthreads();
    if (tid < S_) {
        float m = -3.4e38f;
        for (int s = 0; s < S_; ++s) m = fmaxf(m, sc[s]);
        ex[tid] = expf(sc[tid] - m);
    }
    __syncthreads();
    if (tid == 0) { float d = 0.0f; for (int s = 0; s < S_; ++s) d += ex[s]; denom = d; }
    __syncthreads();
    const float inv = 1.0f / denom;
    for (int u = tid; u < U_; u += 256) {
        float a = 0.0f;
        for (int s = 0; s < S_; ++s) a += ex[s] * o[((long long)b * S_ + s) * U_ + u];
        hc_bf[(long long)b * TU_ + u] = f2bf(a * inv);   // ctx half of [ctx, h]
    }
}

// ---------------------------------------------------------------- logits transpose [t][b][n] -> [b][n][t]
__global__ void __launch_bounds__(256)
transpose_logits(const float* __restrict__ lbuf, float* __restrict__ out) {
    __shared__ float tile[32][33];
    const int bi = blockIdx.x;
    const int b  = bi / (VD_ / 32);
    const int n0 = (bi % (VD_ / 32)) * 32;
    const int tx = threadIdx.x & 31, ty = threadIdx.x >> 5;
#pragma unroll
    for (int j = 0; j < 4; ++j) {
        int tt = ty + j * 8;   // time index
        tile[tt][tx] = lbuf[(long long)tt * B_ * VD_ + (long long)b * VD_ + n0 + tx];
    }
    __syncthreads();
#pragma unroll
    for (int j = 0; j < 4; ++j) {
        int n = ty + j * 8;    // vocab offset in tile
        out[(long long)b * VD_ * T_ + (long long)(n0 + n) * T_ + tx] = tile[tx][n];
    }
}

// ---------------------------------------------------------------- host orchestration
static inline unsigned gemm_blocks(int M, long long N) {
    long long waves = (long long)(M / 16) * (N / 64);
    return (unsigned)((waves + 7) / 8);
}

extern "C" void kernel_launch(void* const* d_in, const int* in_sizes, int n_in,
                              void* d_out, int out_size, void* d_ws, size_t ws_size,
                              hipStream_t stream) {
    (void)in_sizes; (void)n_in; (void)out_size;
    const int*   x       = (const int*)d_in[0];
    const int*   y       = (const int*)d_in[1];
    const float* enc_emb = (const float*)d_in[2];
    const float* dec_emb = (const float*)d_in[3];
    const float* W_ih_e  = (const float*)d_in[4];
    const float* W_hh_e  = (const float*)d_in[5];
    const float* b_ih_e  = (const float*)d_in[6];
    const float* b_hh_e  = (const float*)d_in[7];
    const float* Wa      = (const float*)d_in[8];
    const float* ba      = (const float*)d_in[9];
    const float* W_ih_d  = (const float*)d_in[10];
    const float* W_hh_d  = (const float*)d_in[11];
    const float* b_ih_d  = (const float*)d_in[12];
    const float* b_hh_d  = (const float*)d_in[13];
    const float* Wd      = (const float*)d_in[14];
    const float* bd      = (const float*)d_in[15];
    float* out = (float*)d_out;

    char* ws = (char*)d_ws;
    size_t off = 0;
    auto take = [&](size_t bytes) -> char* {
        char* p = ws + off;
        off = (off + bytes + 255) & ~(size_t)255;
        return p;
    };

    unsigned short* embE = (unsigned short*)take((size_t)B_ * S_ * E_ * 2);
    unsigned short* embD = (unsigned short*)take((size_t)B_ * T_ * E_ * 2);
    unsigned short* wihe = (unsigned short*)take((size_t)FU_ * E_ * 2);
    unsigned short* whhe = (unsigned short*)take((size_t)FU_ * U_ * 2);
    unsigned short* wihd = (unsigned short*)take((size_t)FU_ * E_ * 2);
    unsigned short* whhd = (unsigned short*)take((size_t)FU_ * U_ * 2);
    unsigned short* wab  = (unsigned short*)take((size_t)U_ * U_ * 2);
    unsigned short* wdb  = (unsigned short*)take((size_t)VD_ * TU_ * 2);
    float* zin_e = (float*)take((size_t)B_ * S_ * FU_ * 4);
    float* zin_d = (float*)take((size_t)B_ * T_ * FU_ * 4);
    float* zstep = (float*)take((size_t)B_ * FU_ * 4);
    unsigned short* h_bf  = (unsigned short*)take((size_t)B_ * U_ * 2);
    float* cst  = (float*)take((size_t)B_ * U_ * 4);
    float* obuf = (float*)take((size_t)B_ * S_ * U_ * 4);
    float* qbuf = (float*)take((size_t)B_ * U_ * 4);
    unsigned short* hc_bf = (unsigned short*)take((size_t)B_ * TU_ * 2);

    float* lbuf = nullptr;                        // optional [T][B][VD] staging
    if (off + (size_t)T_ * B_ * VD_ * 4 <= ws_size)
        lbuf = (float*)take((size_t)T_ * B_ * VD_ * 4);

    auto cast = [&](const float* s, unsigned short* d, long long n) {
        castk<<<(unsigned)((n + 255) / 256), 256, 0, stream>>>(s, d, n);
    };
    // Weight casts (bf16 Wd = 131 MB -> L2-resident across all 32 decode steps)
    cast(W_ih_e, wihe, (long long)FU_ * E_);
    cast(W_hh_e, whhe, (long long)FU_ * U_);
    cast(W_ih_d, wihd, (long long)FU_ * E_);
    cast(W_hh_d, whhd, (long long)FU_ * U_);
    cast(Wa,     wab,  (long long)U_ * U_);
    cast(Wd,     wdb,  (long long)VD_ * TU_);

    // Embedding gathers -> bf16
    {
        long long nE = (long long)B_ * S_ * E_;
        gather_emb<<<(unsigned)((nE + 255) / 256), 256, 0, stream>>>(x, S_, S_, enc_emb, embE, nE);
        long long nD = (long long)B_ * T_ * E_;
        gather_emb<<<(unsigned)((nD + 255) / 256), 256, 0, stream>>>(y, T_, T_ + 1, dec_emb, embD, nD);
    }

    // Batched input projections with both biases folded in (WMMA C-init):
    // zin_e[b*S+s, :] = embE @ W_ih_e^T + b_ih_e + b_hh_e
    wmma_gemm_bf16<<<gemm_blocks(B_ * S_, FU_), 256, 0, stream>>>(
        embE, E_, wihe, E_, nullptr, 0, b_ih_e, b_hh_e,
        zin_e, FU_, 1, B_ * S_, FU_, E_);
    wmma_gemm_bf16<<<gemm_blocks(B_ * T_, FU_), 256, 0, stream>>>(
        embD, E_, wihd, E_, nullptr, 0, b_ih_d, b_hh_d,
        zin_d, FU_, 1, B_ * T_, FU_, E_);

    init_state<<<(B_ * U_ + 255) / 256, 256, 0, stream>>>(h_bf, cst, B_ * U_);

    // ---------------- encoder recurrence ----------------
    for (int t = 0; t < S_; ++t) {
        // z = h @ W_hh_e^T + zin_e[:, t, :]   (Z rows strided by S*4U)
        wmma_gemm_bf16<<<gemm_blocks(B_, FU_), 256, 0, stream>>>(
            h_bf, U_, whhe, U_, zin_e + (long long)t * FU_, (long long)S_ * FU_,
            nullptr, nullptr, zstep, FU_, 1, B_, FU_, U_);
        lstm_gates_enc<<<(B_ * U_ + 255) / 256, 256, 0, stream>>>(zstep, cst, h_bf, obuf, t);
    }

    // ---------------- decoder recurrence ----------------
    for (int t = 0; t < T_; ++t) {
        // q = h @ Wa^T + ba
        wmma_gemm_bf16<<<gemm_blocks(B_, U_), 256, 0, stream>>>(
            h_bf, U_, wab, U_, nullptr, 0, ba, nullptr,
            qbuf, U_, 1, B_, U_, U_);
        // attention -> ctx (writes bf16 into hc[:, :U])
        attn_kernel<<<B_, 256, 0, stream>>>(qbuf, obuf, hc_bf);
        // decoder LSTM step
        wmma_gemm_bf16<<<gemm_blocks(B_, FU_), 256, 0, stream>>>(
            h_bf, U_, whhd, U_, zin_d + (long long)t * FU_, (long long)T_ * FU_,
            nullptr, nullptr, zstep, FU_, 1, B_, FU_, U_);
        lstm_gates_dec<<<(B_ * U_ + 255) / 256, 256, 0, stream>>>(zstep, cst, h_bf, hc_bf);
        // logits = [ctx, h] @ Wd^T + bd
        if (lbuf) {
            wmma_gemm_bf16<<<gemm_blocks(B_, VD_), 256, 0, stream>>>(
                hc_bf, TU_, wdb, TU_, nullptr, 0, bd, nullptr,
                lbuf + (long long)t * B_ * VD_, VD_, 1, B_, VD_, TU_);
        } else {
            // direct strided store into out[b, n, t]
            wmma_gemm_bf16<<<gemm_blocks(B_, VD_), 256, 0, stream>>>(
                hc_bf, TU_, wdb, TU_, nullptr, 0, bd, nullptr,
                out + t, (long long)VD_ * T_, T_, B_, VD_, TU_);
        }
    }

    if (lbuf)
        transpose_logits<<<B_ * (VD_ / 32), 256, 0, stream>>>(lbuf, out);
}
